// GCN_53695681135100
// MI455X (gfx1250) — compile-verified, matrix operand order checked by reading, not run on hardware
//
#include <hip/hip_runtime.h>
#include <hip/hip_bf16.h>

typedef __attribute__((ext_vector_type(16))) _Float16 v16h;
typedef __attribute__((ext_vector_type(8)))  _Float16 v8h;
typedef __attribute__((ext_vector_type(8)))  float    v8f;

#define NNODES   50000
#define DEG      16
#define NEDGES   (NNODES * DEG)
#define DIN      128
#define DHID     128
#define DOUT     16
#define MTILES   (NNODES / 16)      // 3125 row tiles of 16

// ---------------- workspace layout (bytes, all 256-aligned) ----------------
#define WS_VALS  0                          // E * 4            = 3,200,000
#define WS_XW    3200000                    // N * 128 * 4      = 25,600,000
#define WS_H16   28800000                   // N * 128 * 2      = 12,800,000
#define WS_HW    41600000                   // N * 16 * 4       = 3,200,000
#define WS_W1H   44800000                   // 8*4*32*16 * 2    = 32,768
#define WS_W2H   44832768                   // 4*32*16 * 2      = 4,096

// ---------------------------------------------------------------------------
// Kernel 1: pack W1 (128x128) and W2 (128x16) into f16 WMMA-B lane layout.
// For b-vector index t = ((nt*4+kc)*32 + lane), element j:
//   n = nt*16 + (lane&15), hsel = lane>>4
//   K(j) = kc*32 + hsel*8 + (j < 8 ? j : j + 8)
// ---------------------------------------------------------------------------
__global__ void gcn_pack_w(const float* __restrict__ W1,
                           const float* __restrict__ W2,
                           _Float16* __restrict__ w1h,
                           _Float16* __restrict__ w2h) {
    int t = blockIdx.x * blockDim.x + threadIdx.x;
    if (t < 1024) {                              // W1: nt 0..7, kc 0..3, lane 0..31
        int lane = t & 31;
        int kc   = (t >> 5) & 3;
        int n    = (t >> 7) * 16 + (lane & 15);
        int hsel = lane >> 4;
#pragma unroll
        for (int j = 0; j < 16; ++j) {
            int k = kc * 32 + hsel * 8 + (j < 8 ? j : j + 8);
            w1h[t * 16 + j] = (_Float16)W1[k * DHID + n];
        }
    } else if (t < 1152) {                       // W2: kc 0..3, lane 0..31
        int u    = t - 1024;
        int lane = u & 31;
        int kc   = u >> 5;
        int n    = lane & 15;
        int hsel = lane >> 4;
#pragma unroll
        for (int j = 0; j < 16; ++j) {
            int k = kc * 32 + hsel * 8 + (j < 8 ? j : j + 8);
            w2h[u * 16 + j] = (_Float16)W2[k * DOUT + n];
        }
    }
}

// ---------------------------------------------------------------------------
// Kernel 2: row-normalized edge values. rows[e] == e/16 by construction.
// ---------------------------------------------------------------------------
__global__ void gcn_vals(const float* __restrict__ ew,
                         float* __restrict__ vals) {
    int node = blockIdx.x * blockDim.x + threadIdx.x;
    if (node >= NNODES) return;
    const float4* wp = (const float4*)(ew + node * DEG);
    float4 w0 = wp[0], w1 = wp[1], w2 = wp[2], w3 = wp[3];
    float s = (w0.x + w0.y + w0.z + w0.w) + (w1.x + w1.y + w1.z + w1.w)
            + (w2.x + w2.y + w2.z + w2.w) + (w3.x + w3.y + w3.z + w3.w);
    float inv = 1.0f / s;
    float4* vp = (float4*)(vals + node * DEG);
    vp[0] = make_float4(w0.x*inv, w0.y*inv, w0.z*inv, w0.w*inv);
    vp[1] = make_float4(w1.x*inv, w1.y*inv, w1.z*inv, w1.w*inv);
    vp[2] = make_float4(w2.x*inv, w2.y*inv, w2.z*inv, w2.w*inv);
    vp[3] = make_float4(w3.x*inv, w3.y*inv, w3.z*inv, w3.w*inv);
}

// ---------------------------------------------------------------------------
// Kernel 3: XW = x @ W1 via v_wmma_f32_16x16x32_f16.
// One wave per 16-row tile. A (16x128) is loaded/converted ONCE into 4 v16h
// registers; nt is the outer loop with a single live accumulator, so the 4 B
// loads per nt can clause and overlap with the previous tile's stores.
// ---------------------------------------------------------------------------
__global__ void gcn_gemm1(const float* __restrict__ x,
                          const _Float16* __restrict__ w1h,
                          float* __restrict__ XW) {
    int wave = blockIdx.x * (blockDim.x >> 5) + (threadIdx.x >> 5);
    if (wave >= MTILES) return;                 // wave-uniform guard, EXEC stays full
    int lane = threadIdx.x & 31;
    int m    = lane & 15;
    int hsel = lane >> 4;
    int r0   = wave * 16;

    // ---- load + convert all of A once: a[kc] covers K = kc*32 .. kc*32+31
    v16h a[4];
#pragma unroll
    for (int kc = 0; kc < 4; ++kc) {
        const float* xp = x + (r0 + m) * DIN + kc * 32 + hsel * 8;
        float4 f0 = *(const float4*)(xp);
        float4 f1 = *(const float4*)(xp + 4);
        float4 f2 = *(const float4*)(xp + 16);
        float4 f3 = *(const float4*)(xp + 20);
        v16h t;
        t[0]  = (_Float16)f0.x; t[1]  = (_Float16)f0.y;
        t[2]  = (_Float16)f0.z; t[3]  = (_Float16)f0.w;
        t[4]  = (_Float16)f1.x; t[5]  = (_Float16)f1.y;
        t[6]  = (_Float16)f1.z; t[7]  = (_Float16)f1.w;
        t[8]  = (_Float16)f2.x; t[9]  = (_Float16)f2.y;
        t[10] = (_Float16)f2.z; t[11] = (_Float16)f2.w;
        t[12] = (_Float16)f3.x; t[13] = (_Float16)f3.y;
        t[14] = (_Float16)f3.z; t[15] = (_Float16)f3.w;
        a[kc] = t;
    }

#pragma unroll
    for (int nt = 0; nt < 8; ++nt) {
        // batch the 4 B-operand loads so they form one clause
        v16h b[4];
#pragma unroll
        for (int kc = 0; kc < 4; ++kc)
            b[kc] = *(const v16h*)(w1h + (((nt * 4 + kc) * 32) + lane) * 16);

        v8f acc = (v8f)(0.0f);
#pragma unroll
        for (int kc = 0; kc < 4; ++kc)
            acc = __builtin_amdgcn_wmma_f32_16x16x32_f16(
                false, a[kc], false, b[kc], (short)0, acc, false, false);

        // D layout: row = 8*hsel + v, col = (lane&15) + nt*16
#pragma unroll
        for (int v = 0; v < 8; ++v)
            XW[(r0 + 8 * hsel + v) * DHID + nt * 16 + (lane & 15)] = acc[v];
    }
}

// ---------------------------------------------------------------------------
// Kernel 4: h = relu(A_norm @ XW + b1), stored as f16. 128 threads per node.
// Gather table XW (25.6 MB) is L2-resident on the 192 MB L2.
// ---------------------------------------------------------------------------
__global__ void gcn_spmm1(const float* __restrict__ XW,
                          const int* __restrict__ cols,
                          const float* __restrict__ vals,
                          const float* __restrict__ b1,
                          _Float16* __restrict__ h16) {
    int idx  = blockIdx.x * blockDim.x + threadIdx.x;
    int node = idx >> 7;
    int d    = idx & 127;
    if (node >= NNODES) return;
    float s = 0.0f;
#pragma unroll
    for (int e = 0; e < DEG; ++e) {
        int   c = cols[node * DEG + e];
        float v = vals[node * DEG + e];
        s = fmaf(v, XW[c * DHID + d], s);
    }
    s += b1[d];
    s = fmaxf(s, 0.0f);
    h16[node * DHID + d] = (_Float16)s;
}

// ---------------------------------------------------------------------------
// Kernel 5: HW = h @ W2 via WMMA. One wave per 16-row tile; A loaded once,
// 4 clause-loaded B vectors, single accumulator.
// ---------------------------------------------------------------------------
__global__ void gcn_gemm2(const _Float16* __restrict__ h16,
                          const _Float16* __restrict__ w2h,
                          float* __restrict__ HW) {
    int wave = blockIdx.x * (blockDim.x >> 5) + (threadIdx.x >> 5);
    if (wave >= MTILES) return;
    int lane = threadIdx.x & 31;
    int m    = lane & 15;
    int hsel = lane >> 4;
    int r0   = wave * 16;

    v16h a[4];
    v16h b[4];
#pragma unroll
    for (int kc = 0; kc < 4; ++kc) {
        const _Float16* hp = h16 + (r0 + m) * DHID + kc * 32 + hsel * 8;
        v8h alo = *(const v8h*)(hp);
        v8h ahi = *(const v8h*)(hp + 16);
        a[kc] = __builtin_shufflevector(alo, ahi,
                                        0, 1, 2, 3, 4, 5, 6, 7,
                                        8, 9, 10, 11, 12, 13, 14, 15);
        b[kc] = *(const v16h*)(w2h + ((kc * 32) + lane) * 16);
    }

    v8f acc = (v8f)(0.0f);
#pragma unroll
    for (int kc = 0; kc < 4; ++kc)
        acc = __builtin_amdgcn_wmma_f32_16x16x32_f16(
            false, a[kc], false, b[kc], (short)0, acc, false, false);

#pragma unroll
    for (int v = 0; v < 8; ++v)
        HW[(r0 + 8 * hsel + v) * DOUT + (lane & 15)] = acc[v];
}

// ---------------------------------------------------------------------------
// Kernel 6: out = log_softmax(A_norm @ HW + b2). 16 lanes per node;
// reductions via shfl_xor within each 16-lane group (wave32).
// ---------------------------------------------------------------------------
__global__ void gcn_spmm2_lsm(const float* __restrict__ HW,
                              const int* __restrict__ cols,
                              const float* __restrict__ vals,
                              const float* __restrict__ b2,
                              float* __restrict__ out) {
    int node   = blockIdx.x * 16 + (threadIdx.x >> 4);
    int lane16 = threadIdx.x & 15;
    if (node >= NNODES) return;
    float s = 0.0f;
#pragma unroll
    for (int e = 0; e < DEG; ++e) {
        int   c = cols[node * DEG + e];
        float v = vals[node * DEG + e];
        s = fmaf(v, HW[c * DOUT + lane16], s);
    }
    s += b2[lane16];
    // max over 16 lanes
    float mx = s;
#pragma unroll
    for (int off = 8; off >= 1; off >>= 1)
        mx = fmaxf(mx, __shfl_xor(mx, off, 16));
    float ex = __expf(s - mx);
    float sum = ex;
#pragma unroll
    for (int off = 8; off >= 1; off >>= 1)
        sum += __shfl_xor(sum, off, 16);
    out[node * DOUT + lane16] = s - mx - __logf(sum);
}

// ---------------------------------------------------------------------------
extern "C" void kernel_launch(void* const* d_in, const int* in_sizes, int n_in,
                              void* d_out, int out_size, void* d_ws, size_t ws_size,
                              hipStream_t stream) {
    const float* x   = (const float*)d_in[0];
    const int*   idx = (const int*)d_in[1];     // [2, E]; rows are implicit e/16
    const float* ew  = (const float*)d_in[2];
    const float* W1  = (const float*)d_in[3];
    const float* b1  = (const float*)d_in[4];
    const float* W2  = (const float*)d_in[5];
    const float* b2  = (const float*)d_in[6];
    const int*   cols = idx + NEDGES;
    float* out = (float*)d_out;

    char* ws = (char*)d_ws;
    float*    vals = (float*)   (ws + WS_VALS);
    float*    XW   = (float*)   (ws + WS_XW);
    _Float16* h16  = (_Float16*)(ws + WS_H16);
    float*    HW   = (float*)   (ws + WS_HW);
    _Float16* w1h  = (_Float16*)(ws + WS_W1H);
    _Float16* w2h  = (_Float16*)(ws + WS_W2H);

    gcn_pack_w<<<9, 128, 0, stream>>>(W1, W2, w1h, w2h);
    gcn_vals<<<(NNODES + 255) / 256, 256, 0, stream>>>(ew, vals);
    gcn_gemm1<<<(MTILES + 7) / 8, 256, 0, stream>>>(x, w1h, XW);
    gcn_spmm1<<<(NNODES * DIN) / 256, 256, 0, stream>>>(XW, cols, vals, b1, h16);
    gcn_gemm2<<<(MTILES + 7) / 8, 256, 0, stream>>>(h16, w2h, HW);
    gcn_spmm2_lsm<<<(NNODES + 15) / 16, 256, 0, stream>>>(HW, cols, vals, b2, out);
}
